// CrossAttention_56118042689543
// MI455X (gfx1250) — matrix operand order repaired and checked by reading, not compiled, not run on hardware
//
#include <hip/hip_runtime.h>
#include <hip/hip_bf16.h>

namespace {

constexpr int kB  = 2;
constexpr int kNQ = 2048;
constexpr int kNK = 4096;
constexpr int kD  = 1024;
constexpr int kH  = 16;
constexpr int kHD = 64;
constexpr float kScale = 0.125f;            // HD^-0.5
constexpr float kLog2e = 1.44269504088896f;
constexpr float kNeg   = -1.0e20f;

typedef __attribute__((ext_vector_type(16))) __bf16 v16bf;
typedef __attribute__((ext_vector_type(8)))  float  v8f;
typedef __attribute__((ext_vector_type(4)))  unsigned int v4u;
typedef __attribute__((ext_vector_type(8)))  int v8i;
typedef __attribute__((ext_vector_type(4)))  int v4i;

struct Pack32 { uint4 lo, hi; };

__device__ inline v16bf make_frag(uint4 lo, uint4 hi) {
  Pack32 p{lo, hi};
  return __builtin_bit_cast(v16bf, p);
}

// A-matrix fragment (16x32 bf16). Lane holds row (lane&15); per ISA layout:
//   VGPR0..3 hold K = half*8 + 0..7, VGPR4..7 hold K = 16 + half*8 + 0..7.
template <int RS>
__device__ inline v16bf load_a_frag(const __bf16* base, int row, int k0, int half) {
  const __bf16* p = base + row * RS + k0;
  uint4 lo = *(const uint4*)(p + half * 8);
  uint4 hi = *(const uint4*)(p + 16 + half * 8);
  return make_frag(lo, hi);
}

// B-matrix fragment (32x16 bf16). Lane holds column (lane&15); tile is stored
// N-major (one row per output column), K = half*8 + 0..7 and 16 + half*8 + 0..7.
template <int RS>
__device__ inline v16bf load_b_frag(const __bf16* base, int row, int k0, int half) {
  const __bf16* p = base + row * RS + k0;
  uint4 lo = *(const uint4*)(p + half * 8);
  uint4 hi = *(const uint4*)(p + 16 + half * 8);
  return make_frag(lo, hi);
}

__device__ inline v8f wmma_bf16(v16bf a, v16bf b, v8f c) {
  return __builtin_amdgcn_wmma_f32_16x16x32_bf16(false, a, false, b, (short)0, c,
                                                 false, false);
}

__device__ inline unsigned pack2(float x, float y) {
  unsigned short lo = __builtin_bit_cast(unsigned short, (__bf16)x);
  unsigned short hi = __builtin_bit_cast(unsigned short, (__bf16)y);
  return (unsigned)lo | ((unsigned)hi << 16);
}

__device__ inline float fast_exp2(float x) { return __builtin_amdgcn_exp2f(x); }

__device__ inline unsigned lds_off(const void* p) {
  return (unsigned)(size_t)p;  // LDS aperture: addr[31:0] is the LDS offset
}

// XOR-butterfly max over 16-lane rows using v_permlane16_b32 (VALU, keeps the
// DS pipe free for WMMA fragment loads). Selector nibble i = i ^ M.
constexpr unsigned perm_sel(int m, int base) {
  unsigned v = 0;
  for (int i = 0; i < 8; ++i) v |= (unsigned)(((base + i) ^ m) & 0xF) << (4 * i);
  return v;
}

template <int M>
__device__ inline float xor16_max(float x) {
#if __has_builtin(__builtin_amdgcn_permlane16)
  constexpr unsigned lo = perm_sel(M, 0);
  constexpr unsigned hi = perm_sel(M, 8);
  unsigned u = __builtin_bit_cast(unsigned, x);
  unsigned p = __builtin_amdgcn_permlane16(u, u, lo, hi, false, false);
  return fmaxf(x, __builtin_bit_cast(float, p));
#else
  return fmaxf(x, __shfl_xor(x, M, 16));
#endif
}

// ---------------------------------------------------------------------------
// Tensor Data Mover: 2D bf16 tile, global -> LDS, with LDS row padding of
// 4 DWORDs (8 bf16) after every 32 DWORDs (64 bf16) => padded row stride 72.
// D# fields per CDNA5 ISA.
// ---------------------------------------------------------------------------
__device__ inline void tdm_load_tile_bf16(const void* gptr, unsigned lds_byte_off,
                                          unsigned tile_w, unsigned tile_h,
                                          unsigned row_stride_elems) {
  unsigned long long ga = (unsigned long long)gptr;
  v4u g0;
  g0[0] = 1u;                                        // count=1, user mode
  g0[1] = lds_byte_off;                              // lds_addr (bytes)
  g0[2] = (unsigned)(ga & 0xffffffffull);            // global_addr[31:0]
  g0[3] = (unsigned)((ga >> 32) & 0x01ffffffull)     // global_addr[56:32]
        | (2u << 30);                                // type=2 ("image")

  v8i g1;
  g1[0] = (int)((1u << 16)        // data_size = 1 (2 bytes)
              | (1u << 20)        // pad_enable
              | (4u << 22)        // pad_interval: 32 DWORDs
              | (3u << 25));      // pad_amount:   4 DWORDs
  g1[1] = (int)((tile_w & 0xffffu) << 16);                                // tensor_dim0[15:0]
  g1[2] = (int)(((tile_w >> 16) & 0xffffu) | ((tile_h & 0xffffu) << 16)); // td0[31:16], td1[15:0]
  g1[3] = (int)(((tile_h >> 16) & 0xffffu) | ((tile_w & 0xffffu) << 16)); // td1[31:16], tile_dim0
  g1[4] = (int)(tile_h & 0xffffu);                                        // tile_dim1, tile_dim2=0
  g1[5] = (int)row_stride_elems;                                          // dim0_stride[31:0]
  g1[6] = 0;                                                              // stride[47:32], dim1_stride
  g1[7] = 0;

  v4i z4 = {0, 0, 0, 0};
#if defined(__clang_major__) && (__clang_major__ >= 23)
  v8i z8 = {0, 0, 0, 0, 0, 0, 0, 0};
  __builtin_amdgcn_tensor_load_to_lds(g0, g1, z4, z4, z8, 0);
#else
  __builtin_amdgcn_tensor_load_to_lds(g0, g1, z4, z4, 0);
#endif
}

// ---------------------------------------------------------------------------
// C[M,N] = A[M,K] * W[N,K]^T (+ bias), bf16 WMMA with fp32 accumulate.
// 256 threads = 8 waves; 128x128 block tile; wave tile 64x32 (4x2 WMMA tiles).
// ---------------------------------------------------------------------------
template <typename AT, bool HAS_BIAS, typename OT>
__global__ __launch_bounds__(256) void gemm_nt(const AT* __restrict__ A,
                                               const float* __restrict__ W,
                                               const float* __restrict__ bias,
                                               OT* __restrict__ C,
                                               int M, int K, int N) {
  constexpr int RS = 40;  // 32 + 8 pad: 16B-aligned rows, conflict-free b128
  __shared__ __align__(16) __bf16 As[128 * RS];
  __shared__ __align__(16) __bf16 Ws[128 * RS];

  const int tid  = threadIdx.x;
  const int lane = tid & 31;
  const int wave = tid >> 5;
  const int half = lane >> 4;
  const int l16  = lane & 15;
  const int wm   = wave & 1;   // 2 waves along M (64 rows each)
  const int wn   = wave >> 1;  // 4 waves along N (32 cols each)
  const int m0 = blockIdx.y * 128;
  const int n0 = blockIdx.x * 128;

  v8f acc[4][2];
#pragma unroll
  for (int i = 0; i < 4; ++i)
#pragma unroll
    for (int j = 0; j < 2; ++j) acc[i][j] = v8f{};

  for (int kk = 0; kk < K; kk += 32) {
    // Cooperative staging of 128x32 A and W tiles (converted to bf16).
#pragma unroll
    for (int i = 0; i < 4; ++i) {
      int e   = tid + i * 256;   // 1024 chunks of 4 elements
      int row = e >> 3;
      int c   = (e & 7) * 4;
      if constexpr (sizeof(AT) == 4) {
        float4 f = *(const float4*)(A + (size_t)(m0 + row) * K + kk + c);
        uint2 pk;
        pk.x = pack2(f.x, f.y);
        pk.y = pack2(f.z, f.w);
        *(uint2*)&As[row * RS + c] = pk;
      } else {
        *(uint2*)&As[row * RS + c] =
            *(const uint2*)(A + (size_t)(m0 + row) * K + kk + c);
      }
      float4 g = *(const float4*)(W + (size_t)(n0 + row) * K + kk + c);
      uint2 pw;
      pw.x = pack2(g.x, g.y);
      pw.y = pack2(g.z, g.w);
      *(uint2*)&Ws[row * RS + c] = pw;
      if (kk + 32 < K && i == 0) {  // prefetch next K-slice (global_prefetch_b8)
        __builtin_prefetch(A + (size_t)(m0 + row) * K + kk + 32 + c, 0, 1);
        __builtin_prefetch(W + (size_t)(n0 + row) * K + kk + 32 + c, 0, 1);
      }
    }
    __syncthreads();

    v16bf af[4], bfr[2];
#pragma unroll
    for (int mf = 0; mf < 4; ++mf)
      af[mf] = load_a_frag<RS>(As, wm * 64 + mf * 16 + l16, 0, half);
#pragma unroll
    for (int nf = 0; nf < 2; ++nf)
      bfr[nf] = load_b_frag<RS>(Ws, wn * 32 + nf * 16 + l16, 0, half);
#pragma unroll
    for (int mf = 0; mf < 4; ++mf)
#pragma unroll
      for (int nf = 0; nf < 2; ++nf)
        acc[mf][nf] = wmma_bf16(af[mf], bfr[nf], acc[mf][nf]);
    __syncthreads();
  }

#pragma unroll
  for (int nf = 0; nf < 2; ++nf) {
    int col  = n0 + wn * 32 + nf * 16 + l16;
    float bv = HAS_BIAS ? bias[col] : 0.0f;
#pragma unroll
    for (int mf = 0; mf < 4; ++mf) {
#pragma unroll
      for (int r = 0; r < 8; ++r) {
        int row   = m0 + wm * 64 + mf * 16 + r + 8 * half;
        float val = acc[mf][nf][r] + bv;
        if constexpr (sizeof(OT) == 4)
          C[(size_t)row * N + col] = val;
        else
          C[(size_t)row * N + col] = (OT)val;
      }
    }
  }
}

// ---------------------------------------------------------------------------
// Flash attention: grid (NQ/128, H, B); 8 waves x 16 q-rows; 64-key steps.
// Q/K tiles staged by the Tensor Data Mover with double-buffered pipelining
// (issue tile i+1, wait TENSORcnt<=1 so tile i is resident while i+1 flies).
// V is double-buffered too: global loads issued at the top of the iteration,
// transposing LDS stores deferred until after the PV WMMAs.
// Row-sums of P are computed on the matrix core via a ones-fragment; the
// running row-max uses v_permlane16_b32 butterflies (pure VALU).
// ---------------------------------------------------------------------------
__global__ __launch_bounds__(256) void flash_attn(const __bf16* __restrict__ Qh,
                                                  const __bf16* __restrict__ Kh,
                                                  const __bf16* __restrict__ Vh,
                                                  const unsigned char* __restrict__ mask,
                                                  __bf16* __restrict__ Out) {
  constexpr int RSQ = 72;  // 64 + 8 pad (matches TDM pad descriptor)
  constexpr int RSK = 72;
  constexpr int RSV = 72;
  constexpr int RSP = 72;
  __shared__ __align__(16) __bf16 Qs[128 * RSQ];        // [q][d]
  __shared__ __align__(16) __bf16 Ks[2][64 * RSK];      // [kcol][d], double buf
  __shared__ __align__(16) __bf16 Vs[2][64 * RSV];      // [d][kcol], double buf
  __shared__ __align__(16) __bf16 Ps[8][16 * RSP];      // per-wave P staging

  const int tid  = threadIdx.x;
  const int lane = tid & 31;
  const int wave = tid >> 5;
  const int half = lane >> 4;
  const int l16  = lane & 15;
  const int b  = blockIdx.z;
  const int h  = blockIdx.y;
  const int q0 = blockIdx.x * 128;

  const __bf16* Qb = Qh + (size_t)b * kNQ * kD + (size_t)h * kHD;
  const __bf16* Kb = Kh + (size_t)b * kNK * kD + (size_t)h * kHD;
  const __bf16* Vb = Vh + (size_t)b * kNK * kD + (size_t)h * kHD;
  const unsigned char* mb = mask + (size_t)b * kNK;

  const int vrow = tid >> 3;       // 0..31 (V staging: rows vrow, vrow+32)
  const int vcol = (tid & 7) * 8;  // d 0..56

  // Prologue: issue Q (128x64) and K tile 0 (64x64) on the TDM, stage V tile 0.
  if (wave == 0) {
    tdm_load_tile_bf16(Qb + (size_t)q0 * kD, lds_off(Qs), 64u, 128u, (unsigned)kD);
    tdm_load_tile_bf16(Kb, lds_off(&Ks[0][0]), 64u, 64u, (unsigned)kD);
  }
  {
    uint4 va = *(const uint4*)(Vb + (size_t)vrow * kD + vcol);
    uint4 vc = *(const uint4*)(Vb + (size_t)(vrow + 32) * kD + vcol);
    const __bf16* ea = (const __bf16*)&va;
    const __bf16* ec = (const __bf16*)&vc;
#pragma unroll
    for (int j = 0; j < 8; ++j) Vs[0][(vcol + j) * RSV + vrow]      = ea[j];
#pragma unroll
    for (int j = 0; j < 8; ++j) Vs[0][(vcol + j) * RSV + vrow + 32] = ec[j];
  }
  if (wave == 0) __builtin_amdgcn_s_wait_tensorcnt(1);  // Q landed; K0 may fly
  __syncthreads();

  v16bf qa0 = load_a_frag<RSQ>(Qs, wave * 16 + l16, 0, half);
  v16bf qa1 = load_a_frag<RSQ>(Qs, wave * 16 + l16, 32, half);

  // ones B-fragment (bf16 1.0 everywhere) for row-sum-via-WMMA.
  const uint4 ones4 = {0x3f803f80u, 0x3f803f80u, 0x3f803f80u, 0x3f803f80u};
  const v16bf onesb = make_frag(ones4, ones4);

  float mrow[8];
  v8f lacc = v8f{};
  v8f o[4];
#pragma unroll
  for (int r = 0; r < 8; ++r) mrow[r] = -3.0e38f;
#pragma unroll
  for (int dt = 0; dt < 4; ++dt) o[dt] = v8f{};

  const float qk = kScale * kLog2e;  // softmax in exp2 domain

  constexpr int NSTEP = kNK / 64;
  for (int it = 0; it < NSTEP; ++it) {
    const int kt   = it * 64;
    const int cur  = it & 1;
    const int nxt  = cur ^ 1;
    const bool more = (it + 1) < NSTEP;

    // Kick off next tile's staging before touching this tile.
    uint4 va = {0, 0, 0, 0}, vc = {0, 0, 0, 0};
    if (more) {
      if (wave == 0)
        tdm_load_tile_bf16(Kb + (size_t)(kt + 64) * kD, lds_off(&Ks[nxt][0]),
                           64u, 64u, (unsigned)kD);
      va = *(const uint4*)(Vb + (size_t)(kt + 64 + vrow) * kD + vcol);
      vc = *(const uint4*)(Vb + (size_t)(kt + 64 + vrow + 32) * kD + vcol);
    }
    if (wave == 0) {
      if (more) __builtin_amdgcn_s_wait_tensorcnt(1);  // tile `it` resident
      else      __builtin_amdgcn_s_wait_tensorcnt(0);
    }
    __syncthreads();  // (A) release Ks[cur]/Vs[cur]

    const __bf16* kbase = &Ks[cur][0];
    const __bf16* vbase = &Vs[cur][0];

    // Scores: S[16 q x 64 k] as four 16x16 tiles, HD=64 reduced in two chunks.
    v8f s[4];
#pragma unroll
    for (int t = 0; t < 4; ++t) {
      v16bf kb = load_b_frag<RSK>(kbase, t * 16 + l16, 0, half);
      s[t] = wmma_bf16(qa0, kb, v8f{});
      kb = load_b_frag<RSK>(kbase, t * 16 + l16, 32, half);
      s[t] = wmma_bf16(qa1, kb, s[t]);
    }

    bool mv[4];
#pragma unroll
    for (int t = 0; t < 4; ++t) mv[t] = mb[kt + t * 16 + l16] != 0;

    float alpha[8];
#pragma unroll
    for (int r = 0; r < 8; ++r) {
      // Mask BEFORE scale (reference order), then scale into log2 domain.
      float x[4];
#pragma unroll
      for (int t = 0; t < 4; ++t) x[t] = (mv[t] ? kNeg : s[t][r]) * qk;
      float mx = fmaxf(fmaxf(x[0], x[1]), fmaxf(x[2], x[3]));
      mx = xor16_max<1>(mx);
      mx = xor16_max<2>(mx);
      mx = xor16_max<4>(mx);
      mx = xor16_max<8>(mx);
      float mn = fmaxf(mrow[r], mx);
      alpha[r] = fast_exp2(mrow[r] - mn);
      mrow[r]  = mn;
#pragma unroll
      for (int t = 0; t < 4; ++t) s[t][r] = fast_exp2(x[t] - mn);
    }

#pragma unroll
    for (int dt = 0; dt < 4; ++dt)
#pragma unroll
      for (int r = 0; r < 8; ++r) o[dt][r] *= alpha[r];
#pragma unroll
    for (int r = 0; r < 8; ++r) lacc[r] *= alpha[r];

    // Bounce P through LDS: C-layout (lane=col) -> A-fragment layout (lane=row).
    __bf16* pw = &Ps[wave][0];
#pragma unroll
    for (int r = 0; r < 8; ++r) {
      int pr = r + 8 * half;
#pragma unroll
      for (int t = 0; t < 4; ++t)
        pw[pr * RSP + t * 16 + l16] = (__bf16)s[t][r];
    }
    __syncthreads();  // (B)

    v16bf pa0 = load_a_frag<RSP>(pw, l16, 0, half);
    v16bf pa1 = load_a_frag<RSP>(pw, l16, 32, half);

    // Row-sums of P on the matrix core: lacc += P * ones.
    lacc = wmma_bf16(pa0, onesb, lacc);
    lacc = wmma_bf16(pa1, onesb, lacc);

#pragma unroll
    for (int dt = 0; dt < 4; ++dt) {
      v16bf vbf = load_b_frag<RSV>(vbase, dt * 16 + l16, 0, half);
      o[dt] = wmma_bf16(pa0, vbf, o[dt]);
      vbf = load_b_frag<RSV>(vbase, dt * 16 + l16, 32, half);
      o[dt] = wmma_bf16(pa1, vbf, o[dt]);
    }

    // Deferred transposing stores of next V tile (loads rode out the compute).
    if (more) {
      const __bf16* ea = (const __bf16*)&va;
      const __bf16* ec = (const __bf16*)&vc;
#pragma unroll
      for (int j = 0; j < 8; ++j) Vs[nxt][(vcol + j) * RSV + vrow]      = ea[j];
#pragma unroll
      for (int j = 0; j < 8; ++j) Vs[nxt][(vcol + j) * RSV + vrow + 32] = ec[j];
    }
    __syncthreads();  // (C) next iteration may overwrite cur buffers
  }

  float invl[8];
#pragma unroll
  for (int r = 0; r < 8; ++r) invl[r] = 1.0f / lacc[r];
#pragma unroll
  for (int dt = 0; dt < 4; ++dt) {
#pragma unroll
    for (int r = 0; r < 8; ++r) {
      int q   = q0 + wave * 16 + r + 8 * half;
      int col = h * kHD + dt * 16 + l16;
      Out[(size_t)(b * kNQ + q) * kD + col] = (__bf16)(o[dt][r] * invl[r]);
    }
  }
}

}  // namespace

extern "C" void kernel_launch(void* const* d_in, const int* in_sizes, int n_in,
                              void* d_out, int out_size, void* d_ws, size_t ws_size,
                              hipStream_t stream) {
  (void)in_sizes; (void)n_in; (void)out_size; (void)ws_size;

  const float* q  = (const float*)d_in[0];
  const float* k  = (const float*)d_in[1];
  const float* v  = (const float*)d_in[2];
  const unsigned char* mask = (const unsigned char*)d_in[3];
  const float* Wq = (const float*)d_in[4];
  const float* Wk = (const float*)d_in[5];
  const float* Wv = (const float*)d_in[6];
  const float* Wp = (const float*)d_in[7];
  const float* bp = (const float*)d_in[8];

  char* ws = (char*)d_ws;
  __bf16* qh = (__bf16*)(ws);
  __bf16* kh = (__bf16*)(ws + (size_t)kB * kNQ * kD * 2);
  __bf16* vh = (__bf16*)(ws + (size_t)kB * kNQ * kD * 2 + (size_t)kB * kNK * kD * 2);
  __bf16* ao = (__bf16*)(ws + (size_t)kB * kNQ * kD * 2 + (size_t)2 * kB * kNK * kD * 2);
  float* out = (float*)d_out;

  dim3 blk(256);
  gemm_nt<float, false, __bf16><<<dim3(kD / 128, (kB * kNQ) / 128), blk, 0, stream>>>(
      q, Wq, nullptr, qh, kB * kNQ, kD, kD);
  gemm_nt<float, false, __bf16><<<dim3(kD / 128, (kB * kNK) / 128), blk, 0, stream>>>(
      k, Wk, nullptr, kh, kB * kNK, kD, kD);
  gemm_nt<float, false, __bf16><<<dim3(kD / 128, (kB * kNK) / 128), blk, 0, stream>>>(
      v, Wv, nullptr, vh, kB * kNK, kD, kD);
  flash_attn<<<dim3(kNQ / 128, kH, kB), blk, 0, stream>>>(qh, kh, vh, mask, ao);
  gemm_nt<__bf16, true, float><<<dim3(kD / 128, (kB * kNQ) / 128), blk, 0, stream>>>(
      ao, Wp, bp, out, kB * kNQ, kD, kD);
}
